// GCN1_87101936763608
// MI455X (gfx1250) — matrix operand order repaired and checked by reading, not compiled, not run on hardware
//
#include <hip/hip_runtime.h>
#include <hip/hip_bf16.h>

// ---------------- model dimensions ----------------
#define NNODES 50000
#define NEDGES 800000
#define NGRAPH 512
#define NTILES_N 3125   // NNODES/16

typedef __attribute__((ext_vector_type(16))) __bf16 v16bf;
typedef __attribute__((ext_vector_type(8)))  float v8f;

// packed-weight fragment offsets (in __bf16 elements; 1 frag = 512 elems = 1KB)
#define FR 512
constexpr size_t PKO_EM1W1 = 0 * FR;      // K16  N64 -> 4 frags
constexpr size_t PKO_EM1W2 = 4 * FR;      // K64  N64 -> 8
constexpr size_t PKO_EM2W1 = 12 * FR;     // 4
constexpr size_t PKO_EM2W2 = 16 * FR;     // 8
constexpr size_t PKO_C1LIN = 24 * FR;     // 3 x (K64 N32 -> 4)
constexpr size_t PKO_C1W1  = 36 * FR;     // 3 x (K32 N64 -> 4)
constexpr size_t PKO_C1W2  = 48 * FR;     // 3 x (K64 N64 -> 8)
constexpr size_t PKO_C2LIN = 72 * FR;     // 3 x 8
constexpr size_t PKO_C2W1  = 96 * FR;     // 3 x 8
constexpr size_t PKO_C2W2  = 120 * FR;    // 3 x 8
constexpr size_t PKO_LIN1  = 144 * FR;    // K192 N64 -> 24
constexpr size_t PKO_LIN2  = 168 * FR;    // 24
constexpr size_t PK_BYTES  = 192 * FR * 2;   // 196608

// workspace byte offsets (all 1KB aligned)
constexpr size_t OFF_AGG1 = PK_BYTES;                       // [N,96] f32
constexpr size_t OFF_H1   = OFF_AGG1 + 19200000;            // [N,64] f32
constexpr size_t OFF_AGG2 = OFF_H1 + 12800000;              // [N,192] f32
constexpr size_t OFF_SUMS = OFF_AGG2 + 38400000;            // [G,64] f32
constexpr size_t OFF_CNT  = OFF_SUMS + 131072;              // [G] f32
constexpr size_t ZERO_FLOATS = (19200000 + 12800000 + 38400000 + 131072 + 2048) / 4;

// ---------------- helpers ----------------
__device__ __forceinline__ __bf16 f2bf(float f) { return (__bf16)f; }

__device__ __forceinline__ v8f v8f_zero() {
  v8f v;
#pragma unroll
  for (int i = 0; i < 8; ++i) v[i] = 0.0f;
  return v;
}

__device__ __forceinline__ v8f wmma_bf16(v16bf a, v16bf b, v8f c) {
  // D = A(16x32 bf16) * B(32x16 bf16) + C(f32)
  return __builtin_amdgcn_wmma_f32_16x16x32_bf16(false, a, false, b, (short)0, c, false, false);
}

// Index of element (row m, column c) inside a tile stored directly in the
// ISA 7.12.2 16-bit A-fragment register layout (bf16 elements):
//   frag f = c>>5 ; kk = c&31 ; lane = m + 16*((kk>>3)&1) ; h = (kk&7) + ((kk>>4)<<3)
__device__ __forceinline__ int pk_idx(int m, int c) {
  int f  = c >> 5;
  int kk = c & 31;
  int hp = (kk >> 3) & 1;
  int h  = (kk & 7) + ((kk >> 4) << 3);
  return f * 512 + (m + (hp << 4)) * 16 + h;
}

// A-fragment load from a pre-packed LDS tile: lane-contiguous 32 bytes
__device__ __forceinline__ v16bf a_frag_pk(const __bf16* t, int frag) {
  return *(const v16bf*)(t + frag * 512 + (threadIdx.x & 31) * 16);
}

// Load prepacked B fragment (lane-contiguous 32B, global)
__device__ __forceinline__ v16bf b_frag(const __bf16* __restrict__ base, int frag) {
  return *(const v16bf*)(base + ((size_t)frag << 9) + ((size_t)(threadIdx.x & 31) << 4));
}

// ---------------- utility kernels ----------------
__global__ void zero_kernel(float* __restrict__ p, long n) {
  long i = (long)blockIdx.x * blockDim.x + threadIdx.x;
  long stride = (long)gridDim.x * blockDim.x;
  for (; i < n; i += stride) p[i] = 0.0f;
}

// Pack W[KxN] (row-major f32) into B-fragment layout: frag = kt*NT + nt,
// element (lane,h): K = kt*32 + (lane>>4)*8 + (h&7) + (h>>3)*16 ; N = nt*16 + (lane&15)
__global__ void pack_b_kernel(const float* __restrict__ W, __bf16* __restrict__ dst,
                              int K, int N) {
  int NT = N >> 4;
  int total = ((K + 31) >> 5) * NT * 512;
  int idx = blockIdx.x * blockDim.x + threadIdx.x;
  if (idx >= total) return;
  int h = idx & 15;
  int lane = (idx >> 4) & 31;
  int frag = idx >> 9;
  int nt = frag % NT, kt = frag / NT;
  int k = kt * 32 + (lane >> 4) * 8 + (h & 7) + ((h >> 3) << 4);
  int n = nt * 16 + (lane & 15);
  float v = (k < K) ? W[(size_t)k * N + n] : 0.0f;
  dst[idx] = f2bf(v);
}

// ---------------- edge kernel, layer 1 ----------------
// per 16-edge tile: t=relu(EA@em1_w1+b1); ea=t@em1_w2+b2;
// for conv k: proj=ea@c1_lin_w[k]; m=relu(x[src]+proj+lin_b); atomicAdd agg1[dst]
__global__ void __launch_bounds__(128) edge_l1_kernel(
    const float* __restrict__ x, const float* __restrict__ edge_attr,
    const int* __restrict__ esrc, const int* __restrict__ edst,
    const float* __restrict__ bm1, const float* __restrict__ bm2,
    const float* __restrict__ lin_b, const __bf16* __restrict__ pk,
    float* __restrict__ agg) {
  __shared__ alignas(32) __bf16 smT[4][1024];  // t tile, packed A layout (K=64)
  __shared__ alignas(32) __bf16 smE[4][1024];  // ea tile, packed A layout
  const int wid = threadIdx.x >> 5, lane = threadIdx.x & 31;
  const int n = lane & 15, hi = lane >> 4;
  const int e0 = (blockIdx.x * 4 + wid) * 16;
  __bf16* Tt = smT[wid];
  __bf16* Te = smE[wid];

  // A frag of edge_attr (K=16 padded to 32)
  v16bf aEA;
  {
    const float* ap = edge_attr + (size_t)(e0 + n) * 16 + hi * 8;
#pragma unroll
    for (int h = 0; h < 8; ++h) aEA[h] = f2bf(ap[h]);
#pragma unroll
    for (int h = 8; h < 16; ++h) aEA[h] = f2bf(0.0f);
  }
#pragma unroll
  for (int nt = 0; nt < 4; ++nt) {
    v8f acc = v8f_zero();
    acc = wmma_bf16(aEA, b_frag(pk + PKO_EM1W1, nt), acc);
    float bias = bm1[nt * 16 + n];
#pragma unroll
    for (int r = 0; r < 8; ++r)
      Tt[pk_idx(r + 8 * hi, nt * 16 + n)] = f2bf(fmaxf(acc[r] + bias, 0.0f));
  }
  __syncthreads();
  v16bf a0 = a_frag_pk(Tt, 0), a1 = a_frag_pk(Tt, 1);
#pragma unroll
  for (int nt = 0; nt < 4; ++nt) {
    v8f acc = v8f_zero();
    acc = wmma_bf16(a0, b_frag(pk + PKO_EM1W2, nt), acc);
    acc = wmma_bf16(a1, b_frag(pk + PKO_EM1W2, 4 + nt), acc);
    float bias = bm2[nt * 16 + n];
#pragma unroll
    for (int r = 0; r < 8; ++r)
      Te[pk_idx(r + 8 * hi, nt * 16 + n)] = f2bf(acc[r] + bias);
  }
  __syncthreads();
  v16bf ae0 = a_frag_pk(Te, 0), ae1 = a_frag_pk(Te, 1);
  for (int k = 0; k < 3; ++k) {
    const __bf16* wb = pk + PKO_C1LIN + (size_t)k * 4 * FR;  // K64 N32: kt*2+nt
#pragma unroll
    for (int nt = 0; nt < 2; ++nt) {
      v8f acc = v8f_zero();
      acc = wmma_bf16(ae0, b_frag(wb, nt), acc);
      acc = wmma_bf16(ae1, b_frag(wb, 2 + nt), acc);
      float lb = lin_b[k * 32 + nt * 16 + n];
#pragma unroll
      for (int r = 0; r < 8; ++r) {
        int e = e0 + r + 8 * hi;
        int s = esrc[e], d = edst[e];
        float xv = x[(size_t)s * 32 + nt * 16 + n];
        float val = fmaxf(acc[r] + lb + xv, 0.0f);
        atomicAdd(&agg[(size_t)d * 96 + k * 32 + nt * 16 + n], val);
      }
    }
  }
}

// ---------------- node kernel, layer 1 ----------------
__global__ void __launch_bounds__(128) node_l1_kernel(
    const float* __restrict__ x, const float* __restrict__ agg1,
    const float* __restrict__ b1, const float* __restrict__ b2,
    const float* __restrict__ lb, const __bf16* __restrict__ pk,
    float* __restrict__ h1) {
  __shared__ alignas(32) __bf16 smH[4][512];   // K=32 tile (1 frag)
  __shared__ alignas(32) __bf16 smT[4][1024];
  __shared__ alignas(32) __bf16 smO[4][1024];
  const int wid = threadIdx.x >> 5, lane = threadIdx.x & 31;
  const int n = lane & 15, hi = lane >> 4;
  int tile = blockIdx.x * 4 + wid;
  if (tile >= NTILES_N) tile = NTILES_N - 1;  // benign recompute
  const int t0 = tile * 16;
  __bf16* H = smH[wid];
  __bf16* T = smT[wid];
  __bf16* O = smO[wid];
  v8f lacc[4];
#pragma unroll
  for (int i = 0; i < 4; ++i) lacc[i] = v8f_zero();

  for (int k = 0; k < 3; ++k) {
    for (int i = lane; i < 512; i += 32) {
      int r = i >> 5, c = i & 31;
      H[pk_idx(r, c)] =
          f2bf(x[(size_t)(t0 + r) * 32 + c] + agg1[(size_t)(t0 + r) * 96 + k * 32 + c]);
    }
    __syncthreads();
    v16bf aH = a_frag_pk(H, 0);
    const __bf16* w1 = pk + PKO_C1W1 + (size_t)k * 4 * FR;  // K32 N64
#pragma unroll
    for (int nt = 0; nt < 4; ++nt) {
      v8f acc = v8f_zero();
      acc = wmma_bf16(aH, b_frag(w1, nt), acc);
      float bias = b1[k * 64 + nt * 16 + n];
#pragma unroll
      for (int r = 0; r < 8; ++r)
        T[pk_idx(r + 8 * hi, nt * 16 + n)] = f2bf(fmaxf(acc[r] + bias, 0.0f));
    }
    __syncthreads();
    v16bf aT0 = a_frag_pk(T, 0), aT1 = a_frag_pk(T, 1);
    const __bf16* w2 = pk + PKO_C1W2 + (size_t)k * 8 * FR;  // K64 N64
#pragma unroll
    for (int nt = 0; nt < 4; ++nt) {
      v8f acc = v8f_zero();
      acc = wmma_bf16(aT0, b_frag(w2, nt), acc);
      acc = wmma_bf16(aT1, b_frag(w2, 4 + nt), acc);
      float bias = b2[k * 64 + nt * 16 + n];
#pragma unroll
      for (int r = 0; r < 8; ++r)
        O[pk_idx(r + 8 * hi, nt * 16 + n)] = f2bf(acc[r] + bias);
    }
    __syncthreads();
    v16bf aO0 = a_frag_pk(O, 0), aO1 = a_frag_pk(O, 1);
#pragma unroll
    for (int nt = 0; nt < 4; ++nt) {
      lacc[nt] = wmma_bf16(aO0, b_frag(pk + PKO_LIN1, (2 * k) * 4 + nt), lacc[nt]);
      lacc[nt] = wmma_bf16(aO1, b_frag(pk + PKO_LIN1, (2 * k + 1) * 4 + nt), lacc[nt]);
    }
    __syncthreads();
  }
#pragma unroll
  for (int nt = 0; nt < 4; ++nt) {
    float bias = lb[nt * 16 + n];
#pragma unroll
    for (int r = 0; r < 8; ++r)
      h1[(size_t)(t0 + r + 8 * hi) * 64 + nt * 16 + n] = fmaxf(lacc[nt][r] + bias, 0.0f);
  }
}

// ---------------- edge kernel, layer 2 ----------------
__global__ void __launch_bounds__(128) edge_l2_kernel(
    const float* __restrict__ h1, const float* __restrict__ edge_attr,
    const int* __restrict__ esrc, const int* __restrict__ edst,
    const float* __restrict__ bm1, const float* __restrict__ bm2,
    const float* __restrict__ lin_b, const __bf16* __restrict__ pk,
    float* __restrict__ agg) {
  __shared__ alignas(32) __bf16 smT[4][1024];
  __shared__ alignas(32) __bf16 smE[4][1024];
  const int wid = threadIdx.x >> 5, lane = threadIdx.x & 31;
  const int n = lane & 15, hi = lane >> 4;
  const int e0 = (blockIdx.x * 4 + wid) * 16;
  __bf16* Tt = smT[wid];
  __bf16* Te = smE[wid];

  v16bf aEA;
  {
    const float* ap = edge_attr + (size_t)(e0 + n) * 16 + hi * 8;
#pragma unroll
    for (int h = 0; h < 8; ++h) aEA[h] = f2bf(ap[h]);
#pragma unroll
    for (int h = 8; h < 16; ++h) aEA[h] = f2bf(0.0f);
  }
#pragma unroll
  for (int nt = 0; nt < 4; ++nt) {
    v8f acc = v8f_zero();
    acc = wmma_bf16(aEA, b_frag(pk + PKO_EM2W1, nt), acc);
    float bias = bm1[nt * 16 + n];
#pragma unroll
    for (int r = 0; r < 8; ++r)
      Tt[pk_idx(r + 8 * hi, nt * 16 + n)] = f2bf(fmaxf(acc[r] + bias, 0.0f));
  }
  __syncthreads();
  v16bf a0 = a_frag_pk(Tt, 0), a1 = a_frag_pk(Tt, 1);
#pragma unroll
  for (int nt = 0; nt < 4; ++nt) {
    v8f acc = v8f_zero();
    acc = wmma_bf16(a0, b_frag(pk + PKO_EM2W2, nt), acc);
    acc = wmma_bf16(a1, b_frag(pk + PKO_EM2W2, 4 + nt), acc);
    float bias = bm2[nt * 16 + n];
#pragma unroll
    for (int r = 0; r < 8; ++r)
      Te[pk_idx(r + 8 * hi, nt * 16 + n)] = f2bf(acc[r] + bias);
  }
  __syncthreads();
  v16bf ae0 = a_frag_pk(Te, 0), ae1 = a_frag_pk(Te, 1);
  for (int k = 0; k < 3; ++k) {
    const __bf16* wb = pk + PKO_C2LIN + (size_t)k * 8 * FR;  // K64 N64: kt*4+nt
#pragma unroll
    for (int nt = 0; nt < 4; ++nt) {
      v8f acc = v8f_zero();
      acc = wmma_bf16(ae0, b_frag(wb, nt), acc);
      acc = wmma_bf16(ae1, b_frag(wb, 4 + nt), acc);
      float lb = lin_b[k * 64 + nt * 16 + n];
#pragma unroll
      for (int r = 0; r < 8; ++r) {
        int e = e0 + r + 8 * hi;
        int s = esrc[e], d = edst[e];
        float xv = h1[(size_t)s * 64 + nt * 16 + n];
        float val = fmaxf(acc[r] + lb + xv, 0.0f);
        atomicAdd(&agg[(size_t)d * 192 + k * 64 + nt * 16 + n], val);
      }
    }
  }
}

// ---------------- node kernel, layer 2 (+ fused mean-pool accumulate) ----------------
__global__ void __launch_bounds__(128) node_l2_kernel(
    const float* __restrict__ h1, const float* __restrict__ agg2,
    const float* __restrict__ b1, const float* __restrict__ b2,
    const float* __restrict__ lb, const int* __restrict__ batch,
    const __bf16* __restrict__ pk, float* __restrict__ sums,
    float* __restrict__ cnt) {
  __shared__ alignas(32) __bf16 smH[4][1024];
  __shared__ alignas(32) __bf16 smT[4][1024];
  __shared__ alignas(32) __bf16 smO[4][1024];
  const int wid = threadIdx.x >> 5, lane = threadIdx.x & 31;
  const int n = lane & 15, hi = lane >> 4;
  const int tileRaw = blockIdx.x * 4 + wid;
  const bool valid = tileRaw < NTILES_N;
  const int t0 = (valid ? tileRaw : NTILES_N - 1) * 16;
  __bf16* H = smH[wid];
  __bf16* T = smT[wid];
  __bf16* O = smO[wid];
  v8f lacc[4];
#pragma unroll
  for (int i = 0; i < 4; ++i) lacc[i] = v8f_zero();

  for (int k = 0; k < 3; ++k) {
    for (int i = lane; i < 1024; i += 32) {
      int r = i >> 6, c = i & 63;
      H[pk_idx(r, c)] =
          f2bf(h1[(size_t)(t0 + r) * 64 + c] + agg2[(size_t)(t0 + r) * 192 + k * 64 + c]);
    }
    __syncthreads();
    v16bf aH0 = a_frag_pk(H, 0), aH1 = a_frag_pk(H, 1);
    const __bf16* w1 = pk + PKO_C2W1 + (size_t)k * 8 * FR;
#pragma unroll
    for (int nt = 0; nt < 4; ++nt) {
      v8f acc = v8f_zero();
      acc = wmma_bf16(aH0, b_frag(w1, nt), acc);
      acc = wmma_bf16(aH1, b_frag(w1, 4 + nt), acc);
      float bias = b1[k * 64 + nt * 16 + n];
#pragma unroll
      for (int r = 0; r < 8; ++r)
        T[pk_idx(r + 8 * hi, nt * 16 + n)] = f2bf(fmaxf(acc[r] + bias, 0.0f));
    }
    __syncthreads();
    v16bf aT0 = a_frag_pk(T, 0), aT1 = a_frag_pk(T, 1);
    const __bf16* w2 = pk + PKO_C2W2 + (size_t)k * 8 * FR;
#pragma unroll
    for (int nt = 0; nt < 4; ++nt) {
      v8f acc = v8f_zero();
      acc = wmma_bf16(aT0, b_frag(w2, nt), acc);
      acc = wmma_bf16(aT1, b_frag(w2, 4 + nt), acc);
      float bias = b2[k * 64 + nt * 16 + n];
#pragma unroll
      for (int r = 0; r < 8; ++r)
        O[pk_idx(r + 8 * hi, nt * 16 + n)] = f2bf(acc[r] + bias);
    }
    __syncthreads();
    v16bf aO0 = a_frag_pk(O, 0), aO1 = a_frag_pk(O, 1);
#pragma unroll
    for (int nt = 0; nt < 4; ++nt) {
      lacc[nt] = wmma_bf16(aO0, b_frag(pk + PKO_LIN2, (2 * k) * 4 + nt), lacc[nt]);
      lacc[nt] = wmma_bf16(aO1, b_frag(pk + PKO_LIN2, (2 * k + 1) * 4 + nt), lacc[nt]);
    }
    __syncthreads();
  }
  // h2 = relu(lacc + lin2_b); pooled sums += h2 per graph
#pragma unroll
  for (int nt = 0; nt < 4; ++nt) {
    float bias = lb[nt * 16 + n];
#pragma unroll
    for (int r = 0; r < 8; ++r) {
      float hv = fmaxf(lacc[nt][r] + bias, 0.0f);
      if (valid) {
        int g = batch[t0 + r + 8 * hi];
        atomicAdd(&sums[(size_t)g * 64 + nt * 16 + n], hv);
      }
    }
  }
  if (valid && lane < 16) {
    int g = batch[t0 + lane];
    atomicAdd(&cnt[g], 1.0f);
  }
}

// ---------------- head ----------------
__global__ void head_kernel(const float* __restrict__ sums, const float* __restrict__ cnt,
                            const float* __restrict__ u, const float* __restrict__ fc_w,
                            const float* __restrict__ fc_b, float* __restrict__ out) {
  int g = blockIdx.x * blockDim.x + threadIdx.x;
  if (g >= NGRAPH) return;
  float c = fmaxf(cnt[g], 1.0f);
  float acc = fc_b[0];
#pragma unroll 4
  for (int f = 0; f < 64; ++f) acc += (sums[g * 64 + f] / c) * fc_w[f];
#pragma unroll 4
  for (int j = 0; j < 32; ++j) acc += u[g * 32 + j] * fc_w[64 + j];
  out[g] = acc;
}

// ---------------- launcher ----------------
extern "C" void kernel_launch(void* const* d_in, const int* in_sizes, int n_in,
                              void* d_out, int out_size, void* d_ws, size_t ws_size,
                              hipStream_t stream) {
  const float* x        = (const float*)d_in[0];
  const float* edge_attr= (const float*)d_in[1];
  const float* u        = (const float*)d_in[2];
  const float* em1_w1   = (const float*)d_in[3];
  const float* em1_b1   = (const float*)d_in[4];
  const float* em1_w2   = (const float*)d_in[5];
  const float* em1_b2   = (const float*)d_in[6];
  const float* em2_w1   = (const float*)d_in[7];
  const float* em2_b1   = (const float*)d_in[8];
  const float* em2_w2   = (const float*)d_in[9];
  const float* em2_b2   = (const float*)d_in[10];
  const float* c1_lin_w = (const float*)d_in[11];
  const float* c1_lin_b = (const float*)d_in[12];
  const float* c1_w1    = (const float*)d_in[13];
  const float* c1_b1    = (const float*)d_in[14];
  const float* c1_w2    = (const float*)d_in[15];
  const float* c1_b2    = (const float*)d_in[16];
  const float* c2_lin_w = (const float*)d_in[17];
  const float* c2_lin_b = (const float*)d_in[18];
  const float* c2_w1    = (const float*)d_in[19];
  const float* c2_b1    = (const float*)d_in[20];
  const float* c2_w2    = (const float*)d_in[21];
  const float* c2_b2    = (const float*)d_in[22];
  const float* lin1_w   = (const float*)d_in[23];
  const float* lin1_b   = (const float*)d_in[24];
  const float* lin2_w   = (const float*)d_in[25];
  const float* lin2_b   = (const float*)d_in[26];
  const float* fc_w     = (const float*)d_in[27];
  const float* fc_b     = (const float*)d_in[28];
  const int* ei         = (const int*)d_in[29];
  const int* batch      = (const int*)d_in[30];
  const int* esrc = ei;
  const int* edst = ei + NEDGES;

  char* ws = (char*)d_ws;
  __bf16* pk  = (__bf16*)ws;
  float* agg1 = (float*)(ws + OFF_AGG1);
  float* h1   = (float*)(ws + OFF_H1);
  float* agg2 = (float*)(ws + OFF_AGG2);
  float* sums = (float*)(ws + OFF_SUMS);
  float* cnt  = (float*)(ws + OFF_CNT);

  zero_kernel<<<2048, 256, 0, stream>>>((float*)(ws + OFF_AGG1), (long)ZERO_FLOATS);

  auto packL = [&](const float* W, size_t fragOffElems, int K, int N) {
    int total = ((K + 31) / 32) * (N / 16) * 512;
    pack_b_kernel<<<(total + 255) / 256, 256, 0, stream>>>(W, pk + fragOffElems, K, N);
  };
  packL(em1_w1, PKO_EM1W1, 16, 64);
  packL(em1_w2, PKO_EM1W2, 64, 64);
  packL(em2_w1, PKO_EM2W1, 16, 64);
  packL(em2_w2, PKO_EM2W2, 64, 64);
  for (int k = 0; k < 3; ++k) {
    packL(c1_lin_w + (size_t)k * 64 * 32, PKO_C1LIN + (size_t)k * 4 * FR, 64, 32);
    packL(c1_w1   + (size_t)k * 32 * 64, PKO_C1W1  + (size_t)k * 4 * FR, 32, 64);
    packL(c1_w2   + (size_t)k * 64 * 64, PKO_C1W2  + (size_t)k * 8 * FR, 64, 64);
    packL(c2_lin_w+ (size_t)k * 64 * 64, PKO_C2LIN + (size_t)k * 8 * FR, 64, 64);
    packL(c2_w1   + (size_t)k * 64 * 64, PKO_C2W1  + (size_t)k * 8 * FR, 64, 64);
    packL(c2_w2   + (size_t)k * 64 * 64, PKO_C2W2  + (size_t)k * 8 * FR, 64, 64);
  }
  packL(lin1_w, PKO_LIN1, 192, 64);
  packL(lin2_w, PKO_LIN2, 192, 64);

  edge_l1_kernel<<<NEDGES / 64, 128, 0, stream>>>(x, edge_attr, esrc, edst,
                                                  em1_b1, em1_b2, c1_lin_b, pk, agg1);
  node_l1_kernel<<<(NTILES_N + 3) / 4, 128, 0, stream>>>(x, agg1, c1_b1, c1_b2, lin1_b, pk, h1);
  edge_l2_kernel<<<NEDGES / 64, 128, 0, stream>>>(h1, edge_attr, esrc, edst,
                                                  em2_b1, em2_b2, c2_lin_b, pk, agg2);
  node_l2_kernel<<<(NTILES_N + 3) / 4, 128, 0, stream>>>(h1, agg2, c2_b1, c2_b2, lin2_b,
                                                         batch, pk, sums, cnt);
  head_kernel<<<(NGRAPH + 255) / 256, 256, 0, stream>>>(sums, cnt, u, fc_w, fc_b, (float*)d_out);
}
